// Linear4bit_44564580663461
// MI455X (gfx1250) — matrix operand order repaired and checked by reading, not compiled, not run on hardware
//
#include <hip/hip_runtime.h>

// ---------------------------------------------------------------------------
// NF4-dequant GEMM for MI455X (gfx1250, wave32, WMMA), double-buffered.
// y[t,o] = sum_k x[t,k] * (codebook[codes[o*K+k]] * absmax[(o*K+k)/64]) + bias[o]
// 128x128 tile / 256-thread block (8 waves); BK=64 (== quant blocksize).
// Pipeline per k-iter: global loads(kt+1) -> 16x v_wmma_f32_16x16x32_bf16(kt)
//                      -> dequant+ds_store(kt+1, other buffer) -> 1 barrier.
// ---------------------------------------------------------------------------

typedef __attribute__((ext_vector_type(16))) __bf16         bf16x16;
typedef __attribute__((ext_vector_type(8)))  float          f32x8;
typedef __attribute__((ext_vector_type(4)))  float          f32x4;
typedef __attribute__((ext_vector_type(4)))  int            i32x4;
typedef __attribute__((ext_vector_type(8)))  unsigned short u16x8;
typedef __attribute__((ext_vector_type(4)))  unsigned short u16x4;

union Frag {               // 32 bytes: one WMMA A/B operand (16 x bf16)
    bf16x16 v;
    u16x8   h[2];
};

#define K_DIM 4096
#define N_DIM 11008
#define BM    128
#define BN    128
#define BK    64           // == quantization blocksize
#define KT    (K_DIM / BK) // 64 k-iterations
#define LDSK  72           // 64 + 8 pad: stride 144B, 16B-aligned, conflict-free rows
#define TILE_ELEMS (BM * LDSK)
// dynamic LDS: 2 x X-tile + 2 x W-tile (bf16) + 16-entry codebook
#define SMEM_BYTES (4 * TILE_ELEMS * 2 + 16 * 4)

__device__ __forceinline__ unsigned short f32_to_bf16(float f) {
    unsigned int u = __float_as_uint(f);
    u += 0x7FFFu + ((u >> 16) & 1u);     // round-to-nearest-even
    return (unsigned short)(u >> 16);
}

__global__ __launch_bounds__(256)
void nf4_linear_wmma_kernel(const float* __restrict__ x,
                            const int*   __restrict__ codes,
                            const float* __restrict__ absmax,
                            const float* __restrict__ codebook,
                            const float* __restrict__ bias,
                            float*       __restrict__ out) {
    extern __shared__ unsigned char smem[];
    unsigned short* ldsX = (unsigned short*)smem;            // [2][BM*LDSK]
    unsigned short* ldsW = ldsX + 2 * TILE_ELEMS;            // [2][BN*LDSK]
    float*          cbS  = (float*)(ldsW + 2 * TILE_ELEMS);  // [16]

    const int tid  = threadIdx.x;
    const int lane = tid & 31;
    const int wave = tid >> 5;
    const int wm   = wave & 3;          // wave row:  rows   wm*32 .. wm*32+31
    const int wn   = wave >> 2;         // wave col:  cols   wn*64 .. wn*64+63
    const int lm   = lane & 15;         // lane-in-half
    const int hh   = lane >> 4;         // lane half (0/1)

    const int mBase = blockIdx.y * BM;
    const int nBase = blockIdx.x * BN;

    if (tid < 16) cbS[tid] = codebook[tid];

    // cooperative loader mapping: 16 float4-columns x 16 base-rows
    const int ldcol = tid & 15;         // float4 / int4 column (k-offset = ldcol*4)
    const int ldrow = tid >> 4;         // base row, stride 16 -> 8 rows per thread

    // -------- software-pipeline registers (raw tile kt+1 in flight) ---------
    f32x4 xr[8];                        // x fp32, 8 rows x 4 k
    i32x4 cr[8];                        // codes,  8 rows x 4 k
    float ar[8];                        // absmax, 8 rows

    auto load_tile = [&](int kt) {      // global -> registers (no waits forced)
        const int k0 = kt * BK;
        #pragma unroll
        for (int i = 0; i < 8; ++i) {
            const int r = ldrow + i * 16;                    // 0..127
            xr[i] = *(const f32x4*)(x + (size_t)(mBase + r) * K_DIM + k0 + ldcol * 4);
            const int o = nBase + r;
            cr[i] = *(const i32x4*)(codes + (size_t)o * K_DIM + k0 + ldcol * 4);
            ar[i] = absmax[o * KT + kt];                     // one quant block per row
        }
    };

    auto store_tile = [&](int kt) {     // dequant + fp32->bf16, registers -> LDS
        unsigned short* dx = ldsX + (kt & 1) * TILE_ELEMS;
        unsigned short* dw = ldsW + (kt & 1) * TILE_ELEMS;
        #pragma unroll
        for (int i = 0; i < 8; ++i) {
            const int r = ldrow + i * 16;
            u16x4 xs;
            xs[0] = f32_to_bf16(xr[i][0]);
            xs[1] = f32_to_bf16(xr[i][1]);
            xs[2] = f32_to_bf16(xr[i][2]);
            xs[3] = f32_to_bf16(xr[i][3]);
            *(u16x4*)&dx[r * LDSK + ldcol * 4] = xs;

            const float am = ar[i];
            u16x4 ws;
            ws[0] = f32_to_bf16(cbS[cr[i][0] & 15] * am);
            ws[1] = f32_to_bf16(cbS[cr[i][1] & 15] * am);
            ws[2] = f32_to_bf16(cbS[cr[i][2] & 15] * am);
            ws[3] = f32_to_bf16(cbS[cr[i][3] & 15] * am);
            *(u16x4*)&dw[r * LDSK + ldcol * 4] = ws;
        }
    };

    f32x8 acc[2][4];
    #pragma unroll
    for (int i = 0; i < 2; ++i)
        #pragma unroll
        for (int j = 0; j < 4; ++j)
            acc[i][j] = f32x8{};

    // -------- prologue: stage tile 0 into buffer 0 --------------------------
    load_tile(0);
    __syncthreads();                    // cbS visible before first dequant
    store_tile(0);
    __syncthreads();

    // -------- main loop: 1 barrier per iteration -----------------------------
    for (int kt = 0; kt < KT; ++kt) {
        const int cur = kt & 1;
        if (kt + 1 < KT) load_tile(kt + 1);      // loads drain under the WMMAs

        const unsigned short* bx = ldsX + cur * TILE_ELEMS;
        const unsigned short* bw = ldsW + cur * TILE_ELEMS;

        #pragma unroll
        for (int ks = 0; ks < 2; ++ks) {                    // two K=32 steps
            Frag a[2], b[4];

            // A (16x32 bf16): lane(l%16)=row M; K = {h*8+0..7, 16+h*8+0..7}
            #pragma unroll
            for (int tm = 0; tm < 2; ++tm) {
                const int row = wm * 32 + tm * 16 + lm;
                const unsigned short* p = &bx[row * LDSK + ks * 32 + hh * 8];
                a[tm].h[0] = *(const u16x8*)(p);
                a[tm].h[1] = *(const u16x8*)(p + 16);
            }
            // B (32x16 bf16): lane col N=lane%16; K = {h*8+0..7, 16+h*8+0..7}
            #pragma unroll
            for (int tn = 0; tn < 4; ++tn) {
                const int row = wn * 64 + tn * 16 + lm;     // o-row == B column
                const unsigned short* p = &bw[row * LDSK + ks * 32 + hh * 8];
                b[tn].h[0] = *(const u16x8*)(p);
                b[tn].h[1] = *(const u16x8*)(p + 16);
            }

            #pragma unroll
            for (int tm = 0; tm < 2; ++tm)
                #pragma unroll
                for (int tn = 0; tn < 4; ++tn)
                    acc[tm][tn] = __builtin_amdgcn_wmma_f32_16x16x32_bf16(
                        /*neg_a=*/false, a[tm].v,
                        /*neg_b=*/false, b[tn].v,
                        /*c_mod=*/(short)0, acc[tm][tn],
                        /*reuse_a=*/false, /*reuse_b=*/false);
        }

        if (kt + 1 < KT) store_tile(kt + 1);     // write the *other* buffer
        __syncthreads();                         // single barrier per iteration
    }

    // -------- epilogue: += bias, f32 store -----------------------------------
    float bv[4];
    #pragma unroll
    for (int tn = 0; tn < 4; ++tn)
        bv[tn] = bias[nBase + wn * 64 + tn * 16 + lm];

    #pragma unroll
    for (int tm = 0; tm < 2; ++tm) {
        const int row0 = mBase + wm * 32 + tm * 16 + hh * 8;  // C: VGPR r -> M = r + 8*h
        #pragma unroll
        for (int tn = 0; tn < 4; ++tn) {
            const int col = nBase + wn * 64 + tn * 16 + lm;   // C: N = lane%16
            #pragma unroll
            for (int r = 0; r < 8; ++r)
                out[(size_t)(row0 + r) * N_DIM + col] = acc[tm][tn][r] + bv[tn];
        }
    }
}

extern "C" void kernel_launch(void* const* d_in, const int* in_sizes, int n_in,
                              void* d_out, int out_size, void* d_ws, size_t ws_size,
                              hipStream_t stream) {
    const float* x        = (const float*)d_in[0];   // [tokens, 4096] fp32
    const int*   codes    = (const int*)  d_in[1];   // [11008*4096] int32 in 0..15
    const float* absmax   = (const float*)d_in[2];   // [11008*4096/64]
    const float* codebook = (const float*)d_in[3];   // [16]
    const float* bias     = (const float*)d_in[4];   // [11008]
    float*       out      = (float*)d_out;           // [tokens, 11008] fp32

    const int tokens = in_sizes[0] / K_DIM;          // 8192
    dim3 grid(N_DIM / BN, tokens / BM);              // (86, 64)
    dim3 block(256);                                 // 8 waves (wave32)
    nf4_linear_wmma_kernel<<<grid, block, SMEM_BYTES, stream>>>(x, codes, absmax,
                                                                codebook, bias, out);
}
